// TransformerXLDecoder_78984448573453
// MI455X (gfx1250) — matrix operand order repaired and checked by reading, not compiled, not run on hardware
//
#include <hip/hip_runtime.h>
#include <math.h>

// ---------------------------------------------------------------------------
// Transformer-XL decoder (2 layers, rel-pos attention, banded mask) for
// MI455X / gfx1250.  GEMMs: v_wmma_f32_16x16x32_f16, 128x64 block tile,
// 32x32 per wave (4 WMMAs / K-step), double-buffered LDS (one barrier per
// K-step), A-tile staged with global_load_async_to_lds_b128 (ASYNCcnt path).
// Attention exploits the banded mask (|t-s| <= 50): 101 scores per query
// instead of the dense T x (2T-1) rel matrix.
// ---------------------------------------------------------------------------

#define L_    2
#define B_    4
#define T_    1000
#define C_    768
#define H_    12
#define DK_   64
#define HALFW 50
#define P_    (2 * T_ - 1)   // 1999 relative positions

typedef __attribute__((ext_vector_type(16))) _Float16 v16h;
typedef __attribute__((ext_vector_type(8)))  float    v8f;

union F16Frag { v16h v; uint4 u[2]; };

__device__ __forceinline__ float gelu_tanh(float x) {
  // jax.nn.gelu default (approximate=True, tanh form)
  float x3 = x * x * x;
  return 0.5f * x * (1.0f + tanhf(0.7978845608028654f * (x + 0.044715f * x3)));
}

// --- CDNA5 async global->LDS copy (tracked by ASYNCcnt, see ISA 08) --------
// LDS operand: low 32 bits of a generic pointer into __shared__ are the LDS
// byte offset.
__device__ __forceinline__ void async_copy_b128(const _Float16* gsrc, _Float16* ldst) {
  unsigned laddr = (unsigned)(uintptr_t)ldst;
  asm volatile("global_load_async_to_lds_b128 %0, %1, off"
               :: "v"(laddr), "v"(gsrc) : "memory");
}

__device__ __forceinline__ void wait_async0() {
#if __has_builtin(__builtin_amdgcn_s_wait_asynccnt)
  __builtin_amdgcn_s_wait_asynccnt(0);
#else
  asm volatile("s_wait_asynccnt 0x0" ::: "memory");
#endif
}

// ------------------------------ elementwise --------------------------------

__global__ void scale_kernel(const float* __restrict__ x, float* __restrict__ h,
                             int n, float s) {
  int i = blockIdx.x * blockDim.x + threadIdx.x;
  if (i < n) h[i] = x[i] * s;
}

__global__ void cvt_f32_to_f16(const float* __restrict__ src,
                               _Float16* __restrict__ dst, int n) {
  int i = blockIdx.x * blockDim.x + threadIdx.x;
  if (i < n) dst[i] = (_Float16)src[i];
}

// Rel positional encoding table, written directly in f16 (feeds the Wp GEMM).
__global__ void pe_kernel(_Float16* __restrict__ pe) {
  int i = blockIdx.x * blockDim.x + threadIdx.x;
  if (i < P_ * C_) {
    int r = i / C_;
    int c = i % C_;
    float rel = (float)(T_ - 1 - r);                     // T-1 .. -(T-1)
    float ex  = -(float)(c & ~1) * (9.210340371976184f / (float)C_); // -2j*ln(1e4)/C
    float ang = rel * expf(ex);
    float val = (c & 1) ? cosf(ang) : sinf(ang);
    pe[i] = (_Float16)val;
  }
}

// LayerNorm over C=768, output f16 for the following WMMA GEMMs.
__global__ __launch_bounds__(256)
void ln_kernel(const float* __restrict__ X, const float* __restrict__ gg,
               const float* __restrict__ bb, _Float16* __restrict__ Y, int cols) {
  __shared__ float red[256];
  const int row = blockIdx.x;
  const int tid = threadIdx.x;
  const float* xr = X + (size_t)row * cols;
  float s = 0.f, s2 = 0.f;
  for (int i = tid; i < cols; i += 256) {
    float v = xr[i];
    s += v; s2 += v * v;
  }
  red[tid] = s; __syncthreads();
  for (int o = 128; o > 0; o >>= 1) { if (tid < o) red[tid] += red[tid + o]; __syncthreads(); }
  float mean = red[0] / (float)cols;
  __syncthreads();
  red[tid] = s2; __syncthreads();
  for (int o = 128; o > 0; o >>= 1) { if (tid < o) red[tid] += red[tid + o]; __syncthreads(); }
  float var = red[0] / (float)cols - mean * mean;
  float r = rsqrtf(var + 1e-5f);
  _Float16* yr = Y + (size_t)row * cols;
  for (int i = tid; i < cols; i += 256)
    yr[i] = (_Float16)((xr[i] - mean) * r * gg[i] + bb[i]);
}

// ------------------------------ WMMA GEMM ----------------------------------
// D(MxN) = A(MxK) @ B(KxN) [+ bias] [+ res] [gelu] -> f32 or f16.
// Block tile 128x64 (256 threads / 8 waves; wave = 32x32 = 4 accumulators).
// K-step 32, double-buffered LDS (one barrier per step).
// As[buf]:  128x32 row-major f16 (async global->LDS copies).
// BsT[buf]: 64x32 = B tile transposed (n-major) so a lane's 16 K-values for
//           the CDNA5 B-operand layout are two contiguous b128 LDS loads.
__global__ __launch_bounds__(256)
void wmma_gemm_kernel(const _Float16* __restrict__ A, const _Float16* __restrict__ Bw,
                      const float* __restrict__ bias, const float* __restrict__ res,
                      float* __restrict__ Df, _Float16* __restrict__ Dh,
                      int M, int N, int K, int do_gelu) {
  __shared__ _Float16 As[2 * 128 * 32];
  __shared__ _Float16 BsT[2 * 64 * 32];

  const int t    = threadIdx.x;
  const int lane = t & 31;
  const int wid  = t >> 5;
  const int wm   = wid & 3;       // 4 waves along M (32 rows each)
  const int wn   = wid >> 2;      // 2 waves along N (32 cols each)
  const int lo   = lane & 15;
  const int hi   = lane >> 4;
  const int n0   = blockIdx.x * 64;
  const int m0   = blockIdx.y * 128;

  v8f c00 = {}; v8f c01 = {}; v8f c10 = {}; v8f c11 = {};

  // staging roles
  const int arow = t >> 1;             // 0..127
  const int acol = (t & 1) << 4;       // 0,16  (two b128s per thread)
  const int brow = t >> 3;             // 0..31 (K)
  const int bcol = (t & 7) << 3;       // 0..56
  const int gm   = m0 + arow;
  const int KSTEPS = K >> 5;

  auto stage = [&](int buf, int k0) {
    _Float16* at = &As[buf * (128 * 32)];
    _Float16* bt = &BsT[buf * (64 * 32)];
    // A tile: async global->LDS, zero-pad ragged M rows via ds stores.
    _Float16* adst = at + arow * 32 + acol;
    if (gm < M) {
      const _Float16* asrc = A + (size_t)gm * K + k0 + acol;
      async_copy_b128(asrc,     adst);
      async_copy_b128(asrc + 8, adst + 8);
    } else {
      uint4 z = make_uint4(0u, 0u, 0u, 0u);
      *(uint4*)(adst)     = z;
      *(uint4*)(adst + 8) = z;
    }
    // B tile: load row-major, transpose into BsT[n][k].
    uint4 bv = *(const uint4*)(Bw + (size_t)(k0 + brow) * N + n0 + bcol);
    const _Float16* bp = (const _Float16*)&bv;
#pragma unroll
    for (int e = 0; e < 8; ++e) bt[(bcol + e) * 32 + brow] = bp[e];
    // prefetch B one step further ahead into L2/L0
    if (k0 + 32 < K)
      __builtin_prefetch(Bw + (size_t)(k0 + 32 + brow) * N + n0 + bcol, 0, 1);
  };

  stage(0, 0);

  for (int ks = 0; ks < KSTEPS; ++ks) {
    const int cur = ks & 1;
    wait_async0();        // own async copies into 'cur' complete
    __syncthreads();      // all waves' staging of 'cur' visible (+dscnt flush)
    if (ks + 1 < KSTEPS) stage(cur ^ 1, (ks + 1) << 5);

    const _Float16* at = &As[cur * (128 * 32)];
    const _Float16* bt = &BsT[cur * (64 * 32)];

    // A-frags (ISA layout: lane m=lo holds K {8hi..8hi+7, 16+8hi..23+8hi})
    F16Frag a0, a1, b0, b1;
    const _Float16* a0p = at + (wm * 32 + lo) * 32 + 8 * hi;
    a0.u[0] = *(const uint4*)(a0p);
    a0.u[1] = *(const uint4*)(a0p + 16);
    const _Float16* a1p = at + (wm * 32 + 16 + lo) * 32 + 8 * hi;
    a1.u[0] = *(const uint4*)(a1p);
    a1.u[1] = *(const uint4*)(a1p + 16);
    // B-frags (lane n=lo holds K {8hi..8hi+7, 16+8hi..23+8hi})
    const _Float16* b0p = bt + (wn * 32 + lo) * 32 + 8 * hi;
    b0.u[0] = *(const uint4*)(b0p);
    b0.u[1] = *(const uint4*)(b0p + 16);
    const _Float16* b1p = bt + (wn * 32 + 16 + lo) * 32 + 8 * hi;
    b1.u[0] = *(const uint4*)(b1p);
    b1.u[1] = *(const uint4*)(b1p + 16);

    c00 = __builtin_amdgcn_wmma_f32_16x16x32_f16(false, a0.v, false, b0.v,
                                                 (short)0, c00, false, false);
    c01 = __builtin_amdgcn_wmma_f32_16x16x32_f16(false, a0.v, false, b1.v,
                                                 (short)0, c01, false, false);
    c10 = __builtin_amdgcn_wmma_f32_16x16x32_f16(false, a1.v, false, b0.v,
                                                 (short)0, c10, false, false);
    c11 = __builtin_amdgcn_wmma_f32_16x16x32_f16(false, a1.v, false, b1.v,
                                                 (short)0, c11, false, false);
    // buffer 'cur' is not restaged until after the next iteration's barrier,
    // and WMMA consumed the ds_load data before this wave can reach it.
  }

  // epilogue: C/D layout -> row = g + 8*hi, col = lane&15
#pragma unroll
  for (int rt = 0; rt < 2; ++rt) {
#pragma unroll
    for (int ct = 0; ct < 2; ++ct) {
      v8f cc = rt ? (ct ? c11 : c10) : (ct ? c01 : c00);
      int ncol = n0 + wn * 32 + ct * 16 + lo;
      float badd = bias ? bias[ncol] : 0.0f;
#pragma unroll
      for (int g = 0; g < 8; ++g) {
        int row = m0 + wm * 32 + rt * 16 + hi * 8 + g;
        if (row < M) {
          float val = cc[g] + badd;
          if (res)     val += res[(size_t)row * N + ncol];
          if (do_gelu) val  = gelu_tanh(val);
          if (Dh) Dh[(size_t)row * N + ncol] = (_Float16)val;
          else    Df[(size_t)row * N + ncol] = val;
        }
      }
    }
  }
}

// --------------------------- banded rel-attention --------------------------
// One wave per (b, head, t).  Window s in [t-50, t+50] (exactly the unmasked
// band).  rel_shift resolved analytically: bd[t,s] uses p row (s + T-1 - t).
__global__ __launch_bounds__(32)
void attn_kernel(const float* __restrict__ q, const float* __restrict__ k,
                 const float* __restrict__ v, const float* __restrict__ p,
                 const float* __restrict__ posu, const float* __restrict__ posv,
                 _Float16* __restrict__ o16) {
  const int unit = blockIdx.x;
  const int t  = unit % T_;
  const int bh = unit / T_;
  const int hh = bh % H_;
  const int b  = bh / H_;
  const int lane = threadIdx.x;

  int s_lo = t - HALFW; if (s_lo < 0) s_lo = 0;
  int s_hi = t + HALFW; if (s_hi > T_ - 1) s_hi = T_ - 1;
  const int nS = s_hi - s_lo + 1;   // 51..101

  const int s0 = s_lo + lane, s1 = s0 + 32, s2 = s0 + 64, s3 = s0 + 96;
  const bool g0 = (s0 <= s_hi), g1 = (s1 <= s_hi), g2 = (s2 <= s_hi), g3 = (s3 <= s_hi);

  const float* qrow = q + ((size_t)(b * T_ + t)) * C_ + hh * DK_;
  const float* urow = posu + hh * DK_;
  const float* wrow = posv + hh * DK_;
  const float* kb   = k + ((size_t)b * T_) * C_ + hh * DK_;
  const float* pb   = p + (size_t)(T_ - 1 - t) * C_ + hh * DK_;  // + s*C_ -> row s+T-1-t

  float ac0 = 0.f, ac1 = 0.f, ac2 = 0.f, ac3 = 0.f;
  float bd0 = 0.f, bd1 = 0.f, bd2 = 0.f, bd3 = 0.f;
  for (int d = 0; d < DK_; ++d) {
    float qd = qrow[d];
    float qu = qd + urow[d];
    float qv = qd + wrow[d];
    if (g0) { ac0 += qu * kb[(size_t)s0 * C_ + d]; bd0 += qv * pb[(size_t)s0 * C_ + d]; }
    if (g1) { ac1 += qu * kb[(size_t)s1 * C_ + d]; bd1 += qv * pb[(size_t)s1 * C_ + d]; }
    if (g2) { ac2 += qu * kb[(size_t)s2 * C_ + d]; bd2 += qv * pb[(size_t)s2 * C_ + d]; }
    if (g3) { ac3 += qu * kb[(size_t)s3 * C_ + d]; bd3 += qv * pb[(size_t)s3 * C_ + d]; }
  }

  const float NEG = -3.0e38f;
  float sc0 = g0 ? (ac0 + bd0) * 0.125f : NEG;
  float sc1 = g1 ? (ac1 + bd1) * 0.125f : NEG;
  float sc2 = g2 ? (ac2 + bd2) * 0.125f : NEG;
  float sc3 = g3 ? (ac3 + bd3) * 0.125f : NEG;

  float m = fmaxf(fmaxf(sc0, sc1), fmaxf(sc2, sc3));
  for (int off = 16; off >= 1; off >>= 1) m = fmaxf(m, __shfl_xor(m, off, 32));

  float w0 = g0 ? expf(sc0 - m) : 0.f;
  float w1 = g1 ? expf(sc1 - m) : 0.f;
  float w2 = g2 ? expf(sc2 - m) : 0.f;
  float w3 = g3 ? expf(sc3 - m) : 0.f;
  float sum = w0 + w1 + w2 + w3;
  for (int off = 16; off >= 1; off >>= 1) sum += __shfl_xor(sum, off, 32);
  const float inv = 1.0f / sum;

  // d-parallel V accumulation: lane owns dims {lane, lane+32}
  float o0 = 0.f, o1 = 0.f;
  const float* vb = v + ((size_t)b * T_) * C_ + hh * DK_;
  for (int idx = 0; idx < nS; ++idx) {
    int ch = idx >> 5, src = idx & 31;
    float wsel = (ch == 0) ? w0 : ((ch == 1) ? w1 : ((ch == 2) ? w2 : w3));
    float wv = __shfl(wsel, src, 32);
    const float* vr = vb + (size_t)(s_lo + idx) * C_;
    o0 += wv * vr[lane];
    o1 += wv * vr[lane + 32];
  }
  o0 *= inv; o1 *= inv;

  _Float16* orow = o16 + ((size_t)(b * T_ + t)) * C_ + hh * DK_;
  orow[lane]      = (_Float16)o0;
  orow[lane + 32] = (_Float16)o1;
}

// ------------------------------ launcher -----------------------------------

extern "C" void kernel_launch(void* const* d_in, const int* in_sizes, int n_in,
                              void* d_out, int out_size, void* d_ws, size_t ws_size,
                              hipStream_t stream) {
  (void)in_sizes; (void)n_in; (void)out_size; (void)ws_size;

  const float* x    = (const float*)d_in[0];
  const float* Wq   = (const float*)d_in[1];
  const float* bq   = (const float*)d_in[2];
  const float* Wk   = (const float*)d_in[3];
  const float* bk   = (const float*)d_in[4];
  const float* Wv   = (const float*)d_in[5];
  const float* bv   = (const float*)d_in[6];
  const float* Wo   = (const float*)d_in[7];
  const float* bo   = (const float*)d_in[8];
  const float* Wp   = (const float*)d_in[9];
  const float* posu = (const float*)d_in[10];
  const float* posv = (const float*)d_in[11];
  const float* ln1g = (const float*)d_in[12];
  const float* ln1b = (const float*)d_in[13];
  const float* ln2g = (const float*)d_in[14];
  const float* ln2b = (const float*)d_in[15];
  const float* W1   = (const float*)d_in[16];
  const float* b1   = (const float*)d_in[17];
  const float* W2   = (const float*)d_in[18];
  const float* b2   = (const float*)d_in[19];
  // d_in[20] = att_mask: band |t-s|>50 is hard-coded into attn_kernel.
  float* out = (float*)d_out;

  const int    CC   = C_ * C_;
  const size_t LCC  = (size_t)L_ * CC;
  const size_t NTOK = (size_t)B_ * T_;        // 4000 rows

  // workspace carve-out (~94 MB total)
  char* wsp = (char*)d_ws;
  size_t off = 0;
  auto alloc = [&](size_t bytes) -> void* {
    void* pp = wsp + off;
    off = (off + bytes + 255) & ~(size_t)255;
    return pp;
  };
  float*    h    = (float*)   alloc(NTOK * C_ * sizeof(float));
  _Float16* y16  = (_Float16*)alloc(NTOK * C_ * sizeof(_Float16));
  float*    qb   = (float*)   alloc(NTOK * C_ * sizeof(float));
  float*    kbuf = (float*)   alloc(NTOK * C_ * sizeof(float));
  float*    vbuf = (float*)   alloc(NTOK * C_ * sizeof(float));
  float*    pbuf = (float*)   alloc((size_t)P_ * C_ * sizeof(float));
  _Float16* pe16 = (_Float16*)alloc((size_t)P_ * C_ * sizeof(_Float16));
  _Float16* o16  = (_Float16*)alloc(NTOK * C_ * sizeof(_Float16));
  _Float16* g16  = (_Float16*)alloc(NTOK * C_ * sizeof(_Float16));
  _Float16* w16  = (_Float16*)alloc(7 * LCC * sizeof(_Float16));
  _Float16* Wq16 = w16 + 0 * LCC;
  _Float16* Wk16 = w16 + 1 * LCC;
  _Float16* Wv16 = w16 + 2 * LCC;
  _Float16* Wo16 = w16 + 3 * LCC;
  _Float16* Wp16 = w16 + 4 * LCC;
  _Float16* W116 = w16 + 5 * LCC;
  _Float16* W216 = w16 + 6 * LCC;

  const int nHC = (int)(NTOK * C_);
  const int gEw = (nHC + 255) / 256;
  const int gW  = ((int)LCC + 255) / 256;
  const int gPe = (P_ * C_ + 255) / 256;

  // h = x * sqrt(C)
  scale_kernel<<<gEw, 256, 0, stream>>>(x, h, nHC, sqrtf((float)C_));
  // pos-encoding table (f16)
  pe_kernel<<<gPe, 256, 0, stream>>>(pe16);
  // weights -> f16 (per-call; launcher must be stateless/deterministic)
  cvt_f32_to_f16<<<gW, 256, 0, stream>>>(Wq, Wq16, (int)LCC);
  cvt_f32_to_f16<<<gW, 256, 0, stream>>>(Wk, Wk16, (int)LCC);
  cvt_f32_to_f16<<<gW, 256, 0, stream>>>(Wv, Wv16, (int)LCC);
  cvt_f32_to_f16<<<gW, 256, 0, stream>>>(Wo, Wo16, (int)LCC);
  cvt_f32_to_f16<<<gW, 256, 0, stream>>>(Wp, Wp16, (int)LCC);
  cvt_f32_to_f16<<<gW, 256, 0, stream>>>(W1, W116, (int)LCC);
  cvt_f32_to_f16<<<gW, 256, 0, stream>>>(W2, W216, (int)LCC);

  dim3 gg(C_ / 64, (unsigned)((NTOK + 127) / 128));   // 12 x 32
  dim3 gp(C_ / 64, (unsigned)((P_ + 127) / 128));     // 12 x 16

  for (int l = 0; l < L_; ++l) {
    const size_t wo = (size_t)l * CC;
    // y = LN1(h) -> f16
    ln_kernel<<<(unsigned)NTOK, 256, 0, stream>>>(h, ln1g + l * C_, ln1b + l * C_, y16, C_);
    // q/k/v projections
    wmma_gemm_kernel<<<gg, 256, 0, stream>>>(y16, Wq16 + wo, bq + l * C_, nullptr,
                                             qb, nullptr, (int)NTOK, C_, C_, 0);
    wmma_gemm_kernel<<<gg, 256, 0, stream>>>(y16, Wk16 + wo, bk + l * C_, nullptr,
                                             kbuf, nullptr, (int)NTOK, C_, C_, 0);
    wmma_gemm_kernel<<<gg, 256, 0, stream>>>(y16, Wv16 + wo, bv + l * C_, nullptr,
                                             vbuf, nullptr, (int)NTOK, C_, C_, 0);
    // p = pe @ Wp (no bias)
    wmma_gemm_kernel<<<gp, 256, 0, stream>>>(pe16, Wp16 + wo, nullptr, nullptr,
                                             pbuf, nullptr, P_, C_, C_, 0);
    // banded rel-pos attention -> o (f16)
    attn_kernel<<<B_ * H_ * T_, 32, 0, stream>>>(qb, kbuf, vbuf, pbuf,
                                                 posu + l * H_ * DK_,
                                                 posv + l * H_ * DK_, o16);
    // h = h + o @ Wo + bo   (in-place residual: each element read+written once)
    wmma_gemm_kernel<<<gg, 256, 0, stream>>>(o16, Wo16 + wo, bo + l * C_, h,
                                             h, nullptr, (int)NTOK, C_, C_, 0);
    // y2 = LN2(h) -> f16
    ln_kernel<<<(unsigned)NTOK, 256, 0, stream>>>(h, ln2g + l * C_, ln2b + l * C_, y16, C_);
    // g = gelu(y2 @ W1 + b1) -> f16
    wmma_gemm_kernel<<<gg, 256, 0, stream>>>(y16, W116 + wo, b1 + l * C_, nullptr,
                                             nullptr, g16, (int)NTOK, C_, C_, 1);
    // h = h + g @ W2 + b2  (last layer writes straight to d_out)
    float* dst = (l == L_ - 1) ? out : h;
    wmma_gemm_kernel<<<gg, 256, 0, stream>>>(g16, W216 + wo, b2 + l * C_, h,
                                             dst, nullptr, (int)NTOK, C_, C_, 0);
  }
}
